// MPNNLayer_41429254537630
// MI455X (gfx1250) — compile-verified, matrix operand order checked by reading, not run on hardware
//
#include <hip/hip_runtime.h>
#include <hip/hip_bf16.h>
#include <cstdint>
#include <cstddef>

// ---------------------------------------------------------------------------
// MPNN layer for MI455X (gfx1250): bf16 WMMA GEMMs + fused GELU / scatter /
// LayerNorm.  Wave32, 16x16x32 bf16 WMMA with f32 accumulation.
// K-step 64 (2 WMMA sub-steps per barrier), async global->LDS staging
// (ASYNCcnt-tracked) when the toolchain exposes the gfx1250 builtins.
// ---------------------------------------------------------------------------

typedef __bf16 bf16_t;
typedef __attribute__((ext_vector_type(16))) __bf16 v16bf;
typedef __attribute__((ext_vector_type(8)))  __bf16 v8bf;
typedef __attribute__((ext_vector_type(8)))  float  v8f;
typedef __attribute__((ext_vector_type(4)))  int    v4i;

#define N_NODES   16384
#define E_EDGES   524288
#define H_DIM     256
#define EIN_DIM   512
#define D_DIM     1024
#define LN_EPS    1e-5f
#define INV_SCALE (1.0f / 30.0f)

// LDS tile: 128 rows x 64 K halfwords, row stride 72 halfwords (144 B).
// 144 is a multiple of 16 (ds_load_b128 alignment) and r*36 mod 64 banks is a
// distinct 4-bank group per row -> conflict-free b128 fragment reads.
#define LDS_STRIDE 72

#if defined(__has_builtin)
#if __has_builtin(__builtin_amdgcn_global_load_async_to_lds_b128) && \
    __has_builtin(__builtin_amdgcn_s_wait_asynccnt)
#define USE_ASYNC_LDS 1
#endif
#endif

#ifdef USE_ASYNC_LDS
typedef __attribute__((address_space(1))) v4i g_v4i;   // global int4
typedef __attribute__((address_space(3))) v4i l_v4i;   // LDS int4
__device__ __forceinline__ void stage_b128(const void* g, void* l) {
  __builtin_amdgcn_global_load_async_to_lds_b128((g_v4i*)g, (l_v4i*)l, 0, 0);
}
__device__ __forceinline__ void stage_wait() {
  __builtin_amdgcn_s_wait_asynccnt(0);
}
#else
__device__ __forceinline__ void stage_b128(const void* g, void* l) {
  *(uint4*)l = *(const uint4*)g;
}
__device__ __forceinline__ void stage_wait() {}
#endif

__device__ __forceinline__ float gelu_exact(float x) {
  return 0.5f * x * (1.0f + erff(x * 0.70710678118654752f));
}

// Assemble a 16x16 (x32-K-slice) bf16 WMMA fragment from an LDS tile laid out
// as [row][k] with LDS_STRIDE halfwords per row, k sub-block at `kc`.
// ISA 16-bit A-matrix 16x32 layout: lanes 0-15 hold row=lane, K={0..7,16..23};
// lanes 16-31 hold row=lane-16, K={8..15,24..31}.  B fragments use the same
// pattern applied to B^T (column-major per lane).
__device__ __forceinline__ v16bf load_frag(const bf16_t* s, int row_base,
                                           int kc, int lane) {
  const int r  = row_base + (lane & 15);
  const int hs = (lane >> 4);                 // 0 or 1
  const bf16_t* p = s + r * LDS_STRIDE + kc;
  v8bf lo = *reinterpret_cast<const v8bf*>(p + (hs ? 8  : 0));   // K 0-7  / 8-15
  v8bf hi = *reinterpret_cast<const v8bf*>(p + (hs ? 24 : 16));  // K16-23 / 24-31
  return __builtin_shufflevector(lo, hi, 0, 1, 2, 3, 4, 5, 6, 7,
                                 8, 9, 10, 11, 12, 13, 14, 15);
}

// Epilogue variants
#define EPI_GELU_BF16 0   // C_bf = bf16(gelu(acc + bias))
#define EPI_SCATTER   1   // atomicAdd(dh[src_idx[row]*Nc + col], gelu(acc+bias)/30)
#define EPI_PLAIN_F32 2   // C_f  = acc + bias

// Tiled GEMM: C[M,Nc] = act(A[M,K] * B[K,Nc] + bias), B given pre-transposed
// as bf16 Bt[Nc][K].  Block tile 128x128, 8 waves (wave32), each wave 32x64.
// K must be a multiple of 64.
template <int A_IS_F32, int EPI>
__global__ __launch_bounds__(256)
void gemm_bf16_wmma(const void* __restrict__ Av,
                    const bf16_t* __restrict__ Bt,
                    const float* __restrict__ bias,
                    bf16_t* __restrict__ Cb,
                    float* __restrict__ Cf,
                    const int* __restrict__ src_idx,
                    float* __restrict__ dh,
                    int Nc, int K) {
  __shared__ __align__(16) bf16_t sA[128 * LDS_STRIDE];
  __shared__ __align__(16) bf16_t sB[128 * LDS_STRIDE];

  const int tid  = threadIdx.x;
  const int lane = tid & 31;
  const int wave = tid >> 5;
  const int wm   = wave & 3;   // 4 waves along M (32 rows each)
  const int wn   = wave >> 2;  // 2 waves along N (64 cols each)
  const int m_blk = blockIdx.x * 128;
  const int n_blk = blockIdx.y * 128;

  v8f acc[2][4] = {};

  for (int k0 = 0; k0 < K; k0 += 64) {
    // ---- stage A tile (128 x 64) into LDS as bf16 ----
    if (A_IS_F32) {
      const float* Af = (const float*)Av;
      // branch-free prefetch target: clamp to current tile on the last step
      const int kpf = (k0 + 64 < K) ? (k0 + 64) : k0;
#pragma unroll
      for (int i = 0; i < 8; ++i) {
        int v  = tid + i * 256;          // float4 id, 0..2047
        int r  = v >> 4;                 // 16 float4 per row
        int c4 = (v & 15) * 4;
        const float* gp = Af + (size_t)(m_blk + r) * K + k0 + c4;
        float4 f = *(const float4*)gp;
        __builtin_prefetch(Af + (size_t)(m_blk + r) * K + kpf + c4, 0, 3);
        bf16_t* d = sA + r * LDS_STRIDE + c4;
        d[0] = (bf16_t)f.x; d[1] = (bf16_t)f.y;
        d[2] = (bf16_t)f.z; d[3] = (bf16_t)f.w;
      }
    } else {
      const bf16_t* Ab = (const bf16_t*)Av;
#pragma unroll
      for (int i = 0; i < 4; ++i) {
        int v  = tid + i * 256;          // 8-halfword id, 0..1023
        int r  = v >> 3;                 // 8 per row
        int c8 = (v & 7) * 8;
        stage_b128(Ab + (size_t)(m_blk + r) * K + k0 + c8,
                   sA + r * LDS_STRIDE + c8);
      }
    }
    // ---- stage B^T tile (128 cols x 64 K) ----
#pragma unroll
    for (int i = 0; i < 4; ++i) {
      int v  = tid + i * 256;
      int r  = v >> 3;
      int c8 = (v & 7) * 8;
      stage_b128(Bt + (size_t)(n_blk + r) * K + k0 + c8,
                 sB + r * LDS_STRIDE + c8);
    }
    stage_wait();          // wave's own async copies complete
    __syncthreads();       // all waves' copies visible

    // ---- 16 WMMAs per barrier pair per wave ----
#pragma unroll
    for (int s = 0; s < 2; ++s) {
      const int kc = s * 32;
      v16bf afr[2];
#pragma unroll
      for (int tm = 0; tm < 2; ++tm)
        afr[tm] = load_frag(sA, wm * 32 + tm * 16, kc, lane);
#pragma unroll
      for (int tn = 0; tn < 4; ++tn) {
        v16bf bfr = load_frag(sB, wn * 64 + tn * 16, kc, lane);
#pragma unroll
        for (int tm = 0; tm < 2; ++tm) {
          acc[tm][tn] = __builtin_amdgcn_wmma_f32_16x16x32_bf16(
              false, afr[tm], false, bfr, (short)0, acc[tm][tn], false, false);
        }
      }
    }
    __syncthreads();
  }

  // ---- epilogue ----
  // C/D 16x16 f32 layout: VGPR v, lanes 0-15: (M=v, N=lane); 16-31: (M=v+8).
  const int col_lo = lane & 15;
  const int hs     = lane >> 4;
#pragma unroll
  for (int tm = 0; tm < 2; ++tm) {
#pragma unroll
    for (int tn = 0; tn < 4; ++tn) {
      const int col = n_blk + wn * 64 + tn * 16 + col_lo;
      const float bv = bias[col];
#pragma unroll
      for (int v = 0; v < 8; ++v) {
        const int row = m_blk + wm * 32 + tm * 16 + v + hs * 8;
        float x = acc[tm][tn][v] + bv;
        if (EPI == EPI_GELU_BF16) {
          Cb[(size_t)row * Nc + col] = (bf16_t)gelu_exact(x);
        } else if (EPI == EPI_SCATTER) {
          const int node = src_idx[row];
          atomicAdd(&dh[(size_t)node * Nc + col], gelu_exact(x) * INV_SCALE);
        } else {
          Cf[(size_t)row * Nc + col] = x;
        }
      }
    }
  }
}

// y = LayerNorm(xa + xb) * g + b ; one wave32 per row of 256.
__global__ __launch_bounds__(256)
void add_ln_kernel(const float* __restrict__ xa, const float* __restrict__ xb,
                   const float* __restrict__ g, const float* __restrict__ bvec,
                   float* __restrict__ y_f32, bf16_t* __restrict__ y_bf) {
  const int lane = threadIdx.x & 31;
  const int wave = threadIdx.x >> 5;
  const int row  = blockIdx.x * 8 + wave;
  const size_t base = (size_t)row * H_DIM + lane * 8;

  float4 a0 = *(const float4*)(xa + base);
  float4 a1 = *(const float4*)(xa + base + 4);
  float4 b0 = *(const float4*)(xb + base);
  float4 b1 = *(const float4*)(xb + base + 4);
  float x[8] = {a0.x + b0.x, a0.y + b0.y, a0.z + b0.z, a0.w + b0.w,
                a1.x + b1.x, a1.y + b1.y, a1.z + b1.z, a1.w + b1.w};

  float s = 0.f, ss = 0.f;
#pragma unroll
  for (int i = 0; i < 8; ++i) { s += x[i]; ss += x[i] * x[i]; }
#pragma unroll
  for (int m = 16; m >= 1; m >>= 1) {
    s  += __shfl_xor(s,  m, 32);
    ss += __shfl_xor(ss, m, 32);
  }
  const float mu  = s * (1.0f / H_DIM);
  const float var = ss * (1.0f / H_DIM) - mu * mu;
  const float rs  = rsqrtf(var + LN_EPS);

  const int c = lane * 8;
#pragma unroll
  for (int i = 0; i < 8; ++i) {
    float y = (x[i] - mu) * rs * g[c + i] + bvec[c + i];
    y_f32[base + i] = y;
    if (y_bf) y_bf[base + i] = (bf16_t)y;
  }
}

// fp32 W[K][N] -> bf16 W^T[N][K]
__global__ __launch_bounds__(256)
void wtrans_kernel(const float* __restrict__ W, bf16_t* __restrict__ Wt,
                   int K, int N) {
  size_t idx = (size_t)blockIdx.x * 256 + threadIdx.x;
  if (idx >= (size_t)K * N) return;
  int n = (int)(idx / K);
  int k = (int)(idx % K);
  Wt[idx] = (bf16_t)W[(size_t)k * N + n];
}

__global__ __launch_bounds__(256)
void zero_f32_kernel(float* __restrict__ p, size_t n) {
  size_t idx = (size_t)blockIdx.x * 256 + threadIdx.x;
  if (idx < n) p[idx] = 0.0f;
}

extern "C" void kernel_launch(void* const* d_in, const int* in_sizes, int n_in,
                              void* d_out, int out_size, void* d_ws, size_t ws_size,
                              hipStream_t stream) {
  (void)in_sizes; (void)n_in; (void)out_size; (void)ws_size;

  const float* h_V   = (const float*)d_in[0];
  const float* h_E   = (const float*)d_in[1];
  const int*   e_idx = (const int*)d_in[2];      // src_idx = edge_idx[0] = first E entries
  const float* m0_w  = (const float*)d_in[3];
  const float* m0_b  = (const float*)d_in[4];
  const float* m1_w  = (const float*)d_in[5];
  const float* m1_b  = (const float*)d_in[6];
  const float* m2_w  = (const float*)d_in[7];
  const float* m2_b  = (const float*)d_in[8];
  const float* d0_w  = (const float*)d_in[9];
  const float* d0_b  = (const float*)d_in[10];
  const float* d1_w  = (const float*)d_in[11];
  const float* d1_b  = (const float*)d_in[12];
  const float* d2_w  = (const float*)d_in[13];
  const float* d2_b  = (const float*)d_in[14];
  const float* ln1_g = (const float*)d_in[15];
  const float* ln1_b = (const float*)d_in[16];
  const float* ln2_g = (const float*)d_in[17];
  const float* ln2_b = (const float*)d_in[18];
  float* out = (float*)d_out;

  // ---- workspace carve-up ----
  char* ws = (char*)d_ws;
  size_t off = 0;
  auto carve = [&](size_t bytes) -> char* {
    char* p = ws + off;
    off = (off + bytes + 255) & ~(size_t)255;
    return p;
  };
  bf16_t* wt0   = (bf16_t*)carve((size_t)H_DIM * EIN_DIM * 2);   // [256][512]
  bf16_t* wt1   = (bf16_t*)carve((size_t)H_DIM * H_DIM * 2);     // [256][256]
  bf16_t* wt2   = (bf16_t*)carve((size_t)H_DIM * H_DIM * 2);
  bf16_t* wtd0  = (bf16_t*)carve((size_t)D_DIM * H_DIM * 2);     // [1024][256]
  bf16_t* wtd1  = (bf16_t*)carve((size_t)D_DIM * D_DIM * 2);     // [1024][1024]
  bf16_t* wtd2  = (bf16_t*)carve((size_t)H_DIM * D_DIM * 2);     // [256][1024]
  bf16_t* mbuf0 = (bf16_t*)carve((size_t)E_EDGES * H_DIM * 2);   // 256 MB
  bf16_t* mbuf1 = (bf16_t*)carve((size_t)E_EDGES * H_DIM * 2);   // 256 MB
  float*  dh    = (float*)carve((size_t)N_NODES * H_DIM * 4);
  float*  h_res = (float*)carve((size_t)N_NODES * H_DIM * 4);
  bf16_t* h_bf  = (bf16_t*)carve((size_t)N_NODES * H_DIM * 2);
  bf16_t* dbuf0 = (bf16_t*)carve((size_t)N_NODES * D_DIM * 2);
  bf16_t* dbuf1 = (bf16_t*)carve((size_t)N_NODES * D_DIM * 2);
  float*  dpre  = (float*)carve((size_t)N_NODES * H_DIM * 4);

  // ---- weight transpose + bf16 convert (small, once per launch) ----
  auto wt_blocks = [](size_t n) { return (unsigned)((n + 255) / 256); };
  wtrans_kernel<<<wt_blocks((size_t)EIN_DIM * H_DIM), 256, 0, stream>>>(m0_w, wt0, EIN_DIM, H_DIM);
  wtrans_kernel<<<wt_blocks((size_t)H_DIM * H_DIM),   256, 0, stream>>>(m1_w, wt1, H_DIM, H_DIM);
  wtrans_kernel<<<wt_blocks((size_t)H_DIM * H_DIM),   256, 0, stream>>>(m2_w, wt2, H_DIM, H_DIM);
  wtrans_kernel<<<wt_blocks((size_t)H_DIM * D_DIM),   256, 0, stream>>>(d0_w, wtd0, H_DIM, D_DIM);
  wtrans_kernel<<<wt_blocks((size_t)D_DIM * D_DIM),   256, 0, stream>>>(d1_w, wtd1, D_DIM, D_DIM);
  wtrans_kernel<<<wt_blocks((size_t)D_DIM * H_DIM),   256, 0, stream>>>(d2_w, wtd2, D_DIM, H_DIM);

  // ---- zero scatter target ----
  zero_f32_kernel<<<wt_blocks((size_t)N_NODES * H_DIM), 256, 0, stream>>>(dh, (size_t)N_NODES * H_DIM);

  // ---- edge message MLP: [E,512] -> 256 -> 256 -> scatter ----
  dim3 ge(E_EDGES / 128, H_DIM / 128);   // (4096, 2)
  gemm_bf16_wmma<1, EPI_GELU_BF16><<<ge, 256, 0, stream>>>(
      h_E, wt0, m0_b, mbuf0, nullptr, nullptr, nullptr, H_DIM, EIN_DIM);
  gemm_bf16_wmma<0, EPI_GELU_BF16><<<ge, 256, 0, stream>>>(
      mbuf0, wt1, m1_b, mbuf1, nullptr, nullptr, nullptr, H_DIM, H_DIM);
  gemm_bf16_wmma<0, EPI_SCATTER><<<ge, 256, 0, stream>>>(
      mbuf1, wt2, m2_b, nullptr, nullptr, e_idx, dh, H_DIM, H_DIM);

  // ---- h = LN(h_V + dh) ----
  add_ln_kernel<<<N_NODES / 8, 256, 0, stream>>>(h_V, dh, ln1_g, ln1_b, h_res, h_bf);

  // ---- node MLP: 256 -> 1024 -> 1024 -> 256 ----
  dim3 gd0(N_NODES / 128, D_DIM / 128);  // (128, 8)
  gemm_bf16_wmma<0, EPI_GELU_BF16><<<gd0, 256, 0, stream>>>(
      h_bf, wtd0, d0_b, dbuf0, nullptr, nullptr, nullptr, D_DIM, H_DIM);
  gemm_bf16_wmma<0, EPI_GELU_BF16><<<gd0, 256, 0, stream>>>(
      dbuf0, wtd1, d1_b, dbuf1, nullptr, nullptr, nullptr, D_DIM, D_DIM);
  dim3 gd2(N_NODES / 128, H_DIM / 128);  // (128, 2)
  gemm_bf16_wmma<0, EPI_PLAIN_F32><<<gd2, 256, 0, stream>>>(
      dbuf1, wtd2, d2_b, nullptr, dpre, nullptr, nullptr, H_DIM, D_DIM);

  // ---- out = LN(h + d) ----
  add_ln_kernel<<<N_NODES / 8, 256, 0, stream>>>(h_res, dpre, ln2_g, ln2_b, out, nullptr);
}
